// ConformerLayer_60902636257497
// MI455X (gfx1250) — matrix operand order repaired and checked by reading, not compiled, not run on hardware
//
#include <hip/hip_runtime.h>

// ---------------------------------------------------------------------------
// Conformer layer forward for MI455X (gfx1250):
//   bf16 WMMA (f32 accumulate) + Tensor Data Mover (TDM) LDS staging.
// ---------------------------------------------------------------------------

typedef __attribute__((ext_vector_type(16))) __bf16       v16bf;
typedef __attribute__((ext_vector_type(8)))  float        v8f;
typedef __attribute__((ext_vector_type(4)))  unsigned int v4u;
typedef __attribute__((ext_vector_type(4)))  int          v4i;
typedef __attribute__((ext_vector_type(8)))  int          v8i;

namespace {
constexpr int Tt = 1024;
constexpr int Bb = 8;
constexpr int Dd = 512;
constexpr int Hh = 8;
constexpr int HD = 64;
constexpr int FF = 2048;
constexpr int Mr = Tt * Bb;     // 8192 token rows
constexpr float EPSf = 1e-5f;
}

// --------------------- Tensor Data Mover 2D tile load ----------------------
// Builds a D# (groups 0/1 per CDNA5 ISA 8.3/8.4) for a 2D bf16 tile:
//   tile_d0 elements per row, tile_d1 rows, global row stride stride0 elems,
//   LDS destination padded by pad_amount DWORD-code after every
//   (2 << pad_interval) DWORDs (pad_enable=1).
__device__ __forceinline__ void tdm_load_2d(unsigned lds_addr, const void* gptr,
                                            unsigned tile_d0, unsigned tile_d1,
                                            unsigned long stride0,
                                            unsigned pad_interval,
                                            unsigned pad_amount) {
  const unsigned long ga = (unsigned long)gptr;
  v4u g0;
  g0[0] = 1u;                                        // count=1, user descriptor
  g0[1] = lds_addr;                                  // LDS byte address
  g0[2] = (unsigned)(ga & 0xffffffffu);              // global_addr[31:0]
  g0[3] = (unsigned)((ga >> 32) & 0x01ffffffu)       // global_addr[56:32]
          | (2u << 30);                              // type = 2 ("image")
  v8i g1;
  g1[0] = (int)((1u << 16)                           // data_size = 1 (2 bytes)
                | (1u << 20)                         // pad_enable
                | (pad_interval << 22) | (pad_amount << 25));
  g1[1] = (int)((tile_d0 & 0xffffu) << 16);          // tensor_dim0[15:0]
  g1[2] = (int)((tile_d0 >> 16) | ((tile_d1 & 0xffffu) << 16));  // tensor_dim1
  g1[3] = (int)((tile_d1 >> 16) | (tile_d0 << 16));  // tile_dim0
  g1[4] = (int)(tile_d1 & 0xffffu);                  // tile_dim1 (tile_dim2=0)
  g1[5] = (int)(stride0 & 0xffffffffu);              // tensor_dim0_stride lo
  g1[6] = (int)((stride0 >> 32) & 0xffffu);          // stride hi | dim1_stride=0
  g1[7] = 0;
  v4i z4 = {0, 0, 0, 0};
#if __clang_major__ >= 23
  v8i z8 = {0, 0, 0, 0, 0, 0, 0, 0};
  __builtin_amdgcn_tensor_load_to_lds(g0, g1, z4, z4, z8, 0);
#else
  __builtin_amdgcn_tensor_load_to_lds(g0, g1, z4, z4, 0);
#endif
}

__device__ __forceinline__ unsigned lds_off(const void* p) {
  // LDS flat aperture: low 32 bits of the flat address are the LDS offset.
  return (unsigned)(unsigned long)p;
}

// ---------------------------- f32 -> bf16 copy -----------------------------
__global__ void cvt_bf16_kernel(const float* __restrict__ in,
                                __bf16* __restrict__ out, int n) {
  int i = blockIdx.x * 256 + threadIdx.x;
  if (i < n) out[i] = (__bf16)in[i];
}

// ------------------------------- LayerNorm ---------------------------------
__global__ __launch_bounds__(128)
void ln_kernel(const float* __restrict__ x, const float* __restrict__ g,
               const float* __restrict__ b, float* __restrict__ outF,
               __bf16* __restrict__ outB) {
  const int row = blockIdx.x;
  const int tid = threadIdx.x;
  const float4 v = ((const float4*)(x + (long)row * Dd))[tid];
  float s = v.x + v.y + v.z + v.w;
#pragma unroll
  for (int m = 16; m >= 1; m >>= 1) s += __shfl_xor(s, m, 32);
  __shared__ float red[8];
  const int wv_ = tid >> 5, ln_ = tid & 31;
  if (ln_ == 0) red[wv_] = s;
  __syncthreads();
  const float mu = (red[0] + red[1] + red[2] + red[3]) * (1.0f / Dd);
  const float dx0 = v.x - mu, dx1 = v.y - mu, dx2 = v.z - mu, dx3 = v.w - mu;
  float q = dx0 * dx0 + dx1 * dx1 + dx2 * dx2 + dx3 * dx3;
#pragma unroll
  for (int m = 16; m >= 1; m >>= 1) q += __shfl_xor(q, m, 32);
  if (ln_ == 0) red[4 + wv_] = q;
  __syncthreads();
  const float var = (red[4] + red[5] + red[6] + red[7]) * (1.0f / Dd);
  const float rs = rsqrtf(var + EPSf);
  const float4 gv = ((const float4*)g)[tid];
  const float4 bv = ((const float4*)b)[tid];
  const float o0 = dx0 * rs * gv.x + bv.x;
  const float o1 = dx1 * rs * gv.y + bv.y;
  const float o2 = dx2 * rs * gv.z + bv.z;
  const float o3 = dx3 * rs * gv.w + bv.w;
  if (outF) {
    float4 ov; ov.x = o0; ov.y = o1; ov.z = o2; ov.w = o3;
    ((float4*)(outF + (long)row * Dd))[tid] = ov;
  }
  if (outB) {
    __bf16* ob = outB + (long)row * Dd + tid * 4;
    ob[0] = (__bf16)o0; ob[1] = (__bf16)o1; ob[2] = (__bf16)o2; ob[3] = (__bf16)o3;
  }
}

// ------------------------------- GEMM (WMMA) -------------------------------
// C[M,N] = alpha * act( A[M,K] * W[N,K]^T + bias[N] ) + residual[M,N]
// A, W bf16 row-major. 128x128x32 tiles, 256 threads (8 waves), wave owns a
// 16x128 strip (8 accumulator tiles). Tiles staged by the TDM: wave 0 issues
// tensor_load_to_lds for the next k-tile (with LDS padding 64B->80B rows)
// while all waves run WMMAs on the current one; one barrier per k-step.
template <int ACT>  // 0 = none, 1 = SiLU
__global__ __launch_bounds__(256)
void gemm_kernel(const __bf16* __restrict__ A, const __bf16* __restrict__ W,
                 const float* __restrict__ bias, const float* __restrict__ res,
                 float alpha, float* __restrict__ outF, __bf16* __restrict__ outB,
                 int M, int N, int Kd) {
  constexpr int LDT = 40;  // padded LDS row stride in bf16 (64B data + 16B pad)
  __shared__ __bf16 As[2][128 * LDT];
  __shared__ __bf16 Ws[2][128 * LDT];

  const int tid  = threadIdx.x;
  const int wave = tid >> 5;
  const int lane = tid & 31;
  const int m0 = blockIdx.y * 128;
  const int n0 = blockIdx.x * 128;
  const int l15  = lane & 15;
  const int asel = (lane < 16) ? 0 : 8;    // A-fragment K-chunk select
  const int bsel = (lane < 16) ? 0 : 16;   // unused: B fragment also uses asel
  const int nk = Kd >> 5;

  const unsigned ldsA[2] = {lds_off(&As[0][0]), lds_off(&As[1][0])};
  const unsigned ldsW[2] = {lds_off(&Ws[0][0]), lds_off(&Ws[1][0])};
  auto issue_tiles = [&](int kt, int bsel2) {
    // 128 rows x 32 elems, row stride Kd; pad: every 64B (code 3) add 16B (code 3)
    tdm_load_2d(ldsA[bsel2], A + (long)m0 * Kd + kt * 32, 32u, 128u,
                (unsigned long)Kd, 3u, 3u);
    tdm_load_2d(ldsW[bsel2], W + (long)n0 * Kd + kt * 32, 32u, 128u,
                (unsigned long)Kd, 3u, 3u);
  };

  v8f acc[8];
#pragma unroll
  for (int c = 0; c < 8; ++c)
#pragma unroll
    for (int r = 0; r < 8; ++r) acc[c][r] = 0.0f;

  if (wave == 0) {
    issue_tiles(0, 0);
    __builtin_amdgcn_s_wait_tensorcnt(0);
  }
  __syncthreads();

  int buf = 0;
  for (int kt = 0; kt < nk; ++kt) {
    const bool more = (kt + 1 < nk);
    if (more && wave == 0) issue_tiles(kt + 1, buf ^ 1);

    // A fragment: lane holds row (lane%16), two 8-element K chunks.
    v16bf afrag;
    {
      const __bf16* as = &As[buf][(wave * 16 + l15) * LDT];
      ((uint4*)&afrag)[0] = *(const uint4*)(as + asel);
      ((uint4*)&afrag)[1] = *(const uint4*)(as + 16 + asel);
    }
#pragma unroll
    for (int c = 0; c < 8; ++c) {
      v16bf bfrag;
      const __bf16* ws = &Ws[buf][(c * 16 + l15) * LDT + asel];
      ((uint4*)&bfrag)[0] = *(const uint4*)(ws + 0);
      ((uint4*)&bfrag)[1] = *(const uint4*)(ws + 16);
      acc[c] = __builtin_amdgcn_wmma_f32_16x16x32_bf16(
          false, afrag, false, bfrag, (short)0, acc[c], false, false);
    }
    if (more && wave == 0) __builtin_amdgcn_s_wait_tensorcnt(0);
    __syncthreads();
    buf ^= 1;
  }

  // epilogue: C layout — vgpr r => row r + (lane<16 ? 0 : 8), col = lane%16
  const int rofs = (lane < 16) ? 0 : 8;
#pragma unroll
  for (int c = 0; c < 8; ++c) {
#pragma unroll
    for (int r = 0; r < 8; ++r) {
      const int gm = m0 + wave * 16 + r + rofs;
      const int gn = n0 + c * 16 + l15;
      float v = acc[c][r] + bias[gn];
      if (ACT == 1) v = v / (1.0f + __expf(-v));  // SiLU
      v *= alpha;
      const long off = (long)gm * N + gn;
      if (res)  v += res[off];
      if (outF) outF[off] = v;
      if (outB) outB[off] = (__bf16)v;
    }
  }
}

// -------------------------- RoPE + repack for attention --------------------
__global__ void rope_pack_kernel(const float* __restrict__ qf,
                                 const float* __restrict__ kf,
                                 const float* __restrict__ vf,
                                 __bf16* __restrict__ qb,
                                 __bf16* __restrict__ kb,
                                 __bf16* __restrict__ vt) {
  const long idx = (long)blockIdx.x * 256 + threadIdx.x;  // T*B*H*32
  const int i = (int)(idx & 31);
  const int h = (int)((idx >> 5) & 7);
  const int b = (int)((idx >> 8) & 7);
  const int t = (int)(idx >> 11);
  const long src = (((long)t * Bb + b) * Hh + h) * HD;
  const float inv = __expf(-(float)i * (1.0f / 32.0f) * 9.210340371976184f);
  const float ang = (float)t * inv;
  float s, c;
  __sincosf(ang, &s, &c);
  const float q0 = qf[src + i], q1 = qf[src + 32 + i];
  const float k0 = kf[src + i], k1 = kf[src + 32 + i];
  const float v0 = vf[src + i], v1 = vf[src + 32 + i];
  const float qr0 = (q0 * c - q1 * s) * 0.125f;
  const float qr1 = (q1 * c + q0 * s) * 0.125f;
  const float kr0 = k0 * c - k1 * s;
  const float kr1 = k1 * c + k0 * s;
  const long bh = (long)b * Hh + h;
  const long dq = (bh * Tt + t) * HD;
  qb[dq + i]      = (__bf16)qr0;
  qb[dq + 32 + i] = (__bf16)qr1;
  kb[dq + i]      = (__bf16)kr0;
  kb[dq + 32 + i] = (__bf16)kr1;
  vt[(bh * HD + i) * Tt + t]      = (__bf16)v0;
  vt[(bh * HD + 32 + i) * Tt + t] = (__bf16)v1;
}

// ----------------------- Flash-style attention (WMMA) ----------------------
// Block = 256 threads (8 waves): one (b,h) head, 128 query rows (16 per wave).
// K / V^T tiles DMA'd into LDS by the TDM (padded rows); online softmax in
// C-layout registers; P converted to A-fragment layout via a per-wave LDS slab.
__global__ __launch_bounds__(256)
void attn_kernel(const __bf16* __restrict__ qb, const __bf16* __restrict__ kb,
                 const __bf16* __restrict__ vt, __bf16* __restrict__ ctx) {
  constexpr int LK = 72;   // Ks row stride (bf16): 128B data + 16B pad
  constexpr int LV = 136;  // Vts row stride: 256B data + 16B pad
  constexpr int LP = 40;   // P staging row stride
  __shared__ __bf16 Ks[128 * LK];
  __shared__ __bf16 Vts[64 * LV];
  __shared__ __bf16 Ps[8][16 * LP];

  const int tid  = threadIdx.x;
  const int wave = tid >> 5;
  const int lane = tid & 31;
  const int l15  = lane & 15;
  const int asel = (lane < 16) ? 0 : 8;
  const int bsel = (lane < 16) ? 0 : 16;
  const int rofs = (lane < 16) ? 0 : 8;

  const int bh = blockIdx.y;
  const int q0row = blockIdx.x * 128 + wave * 16;
  const __bf16* qbase = qb + (long)bh * Tt * HD;
  const __bf16* kbase = kb + (long)bh * Tt * HD;
  const __bf16* vbase = vt + (long)bh * HD * Tt;
  const unsigned ldsK = lds_off(&Ks[0]);
  const unsigned ldsV = lds_off(&Vts[0]);

  // Q fragments (reused across all kv tiles): two k-steps of 32 over Hd=64.
  v16bf qa[2];
#pragma unroll
  for (int ks = 0; ks < 2; ++ks) {
    const __bf16* p = qbase + (long)(q0row + l15) * HD + ks * 32;
    ((uint4*)&qa[ks])[0] = *(const uint4*)(p + asel);
    ((uint4*)&qa[ks])[1] = *(const uint4*)(p + 16 + asel);
  }

  float mi[8], li[8];
  v8f o[4];
#pragma unroll
  for (int r = 0; r < 8; ++r) { mi[r] = -3.0e38f; li[r] = 0.0f; }
#pragma unroll
  for (int ct = 0; ct < 4; ++ct)
#pragma unroll
    for (int r = 0; r < 8; ++r) o[ct][r] = 0.0f;

  for (int j = 0; j < 8; ++j) {  // 8 kv tiles of 128 rows
    __syncthreads();  // all waves done reading the previous tile
    if (wave == 0) {
      // K tile: 128 rows x 64 elems, stride 64; pad 16B every 128B (codes 4,3)
      tdm_load_2d(ldsK, kbase + (long)j * 128 * HD, 64u, 128u,
                  (unsigned long)HD, 4u, 3u);
      // V^T tile: 64 rows x 128 elems, stride T; pad 16B every 256B (codes 5,3)
      tdm_load_2d(ldsV, vbase + j * 128, 128u, 64u,
                  (unsigned long)Tt, 5u, 3u);
      __builtin_amdgcn_s_wait_tensorcnt(0);
    }
    __syncthreads();

    // S = Q * K^T over this tile: 8 C tiles of 16x16
    v8f s[8];
#pragma unroll
    for (int c = 0; c < 8; ++c)
#pragma unroll
      for (int r = 0; r < 8; ++r) s[c][r] = 0.0f;
#pragma unroll
    for (int ks = 0; ks < 2; ++ks) {
#pragma unroll
      for (int c = 0; c < 8; ++c) {
        v16bf bfrag;
        const __bf16* ws = &Ks[(c * 16 + l15) * LK + ks * 32 + asel];
        ((uint4*)&bfrag)[0] = *(const uint4*)(ws + 0);
        ((uint4*)&bfrag)[1] = *(const uint4*)(ws + 16);
        s[c] = __builtin_amdgcn_wmma_f32_16x16x32_bf16(
            false, qa[ks], false, bfrag, (short)0, s[c], false, false);
      }
    }

    // online softmax: row reductions across lane groups of 16
    float mx[8];
#pragma unroll
    for (int r = 0; r < 8; ++r) {
      float m = s[0][r];
#pragma unroll
      for (int c = 1; c < 8; ++c) m = fmaxf(m, s[c][r]);
      mx[r] = m;
    }
#pragma unroll
    for (int m = 8; m >= 1; m >>= 1)
#pragma unroll
      for (int r = 0; r < 8; ++r) mx[r] = fmaxf(mx[r], __shfl_xor(mx[r], m, 32));
    float sc[8];
#pragma unroll
    for (int r = 0; r < 8; ++r) {
      const float mn = fmaxf(mi[r], mx[r]);
      sc[r] = __expf(mi[r] - mn);
      mi[r] = mn;
    }
#pragma unroll
    for (int ct = 0; ct < 4; ++ct)
#pragma unroll
      for (int r = 0; r < 8; ++r) o[ct][r] *= sc[r];
    float ps[8];
#pragma unroll
    for (int r = 0; r < 8; ++r) ps[r] = 0.0f;
#pragma unroll
    for (int c = 0; c < 8; ++c)
#pragma unroll
      for (int r = 0; r < 8; ++r) {
        const float e = __expf(s[c][r] - mi[r]);
        s[c][r] = e;
        ps[r] += e;
      }
#pragma unroll
    for (int m = 8; m >= 1; m >>= 1)
#pragma unroll
      for (int r = 0; r < 8; ++r) ps[r] += __shfl_xor(ps[r], m, 32);
#pragma unroll
    for (int r = 0; r < 8; ++r) li[r] = li[r] * sc[r] + ps[r];

    // O += P * V : convert P (C layout) -> A fragment via per-wave LDS slab
#pragma unroll
    for (int p2 = 0; p2 < 4; ++p2) {
#pragma unroll
      for (int cc = 0; cc < 2; ++cc) {
        const int c = 2 * p2 + cc;
#pragma unroll
        for (int r = 0; r < 8; ++r)
          Ps[wave][(r + rofs) * LP + cc * 16 + l15] = (__bf16)s[c][r];
      }
      v16bf pa;
      {
        const __bf16* pp = &Ps[wave][l15 * LP];
        ((uint4*)&pa)[0] = *(const uint4*)(pp + asel);
        ((uint4*)&pa)[1] = *(const uint4*)(pp + 16 + asel);
      }
#pragma unroll
      for (int ct = 0; ct < 4; ++ct) {
        v16bf vb;
        const __bf16* vp = &Vts[(ct * 16 + l15) * LV + p2 * 32 + asel];
        ((uint4*)&vb)[0] = *(const uint4*)(vp + 0);
        ((uint4*)&vb)[1] = *(const uint4*)(vp + 16);
        o[ct] = __builtin_amdgcn_wmma_f32_16x16x32_bf16(
            false, pa, false, vb, (short)0, o[ct], false, false);
      }
    }
  }

  // write ctx bf16 in [T,B,D] layout (feeds the O-projection GEMM)
  const int bat = bh >> 3, h = bh & 7;
#pragma unroll
  for (int r = 0; r < 8; ++r) {
    const float inv = 1.0f / li[r];
    const int t = q0row + r + rofs;
#pragma unroll
    for (int ct = 0; ct < 4; ++ct) {
      const int hd = ct * 16 + l15;
      ctx[((long)t * Bb + bat) * Dd + h * 64 + hd] = (__bf16)(o[ct][r] * inv);
    }
  }
}

// --------------------------------- GLU -------------------------------------
__global__ void glu_kernel(const float* __restrict__ pw1, float* __restrict__ out) {
  const long i = (long)blockIdx.x * 256 + threadIdx.x;  // M*D
  const long m = i >> 9;
  const int d = (int)(i & 511);
  const float a = pw1[m * 1024 + d];
  const float g = pw1[m * 1024 + 512 + d];
  out[i] = a / (1.0f + __expf(-g));
}

// --------------------------- Depthwise conv K=31 ---------------------------
__global__ void dwconv_kernel(const float* __restrict__ in,
                              const float* __restrict__ w,
                              const float* __restrict__ bias,
                              float* __restrict__ out) {
  const long i = (long)blockIdx.x * 256 + threadIdx.x;  // [t][b][d]
  const int d = (int)(i & 511);
  const long tb = i >> 9;
  const int b = (int)(tb & 7);
  const int t = (int)(tb >> 3);
  // pull the tail of the sliding window toward the caches (global_prefetch_b8)
  const int tp = (t + 16 < Tt) ? (t + 16) : (Tt - 1);
  __builtin_prefetch(&in[(((long)tp * Bb + b) << 9) + d], 0, 1);
  float acc = bias[d];
#pragma unroll
  for (int k = 0; k < 31; ++k) {
    const int ts = t + k - 15;
    if (ts >= 0 && ts < Tt) acc += in[(((long)ts * Bb + b) << 9) + d] * w[d * 31 + k];
  }
  out[i] = acc;
}

// ----------------------- Per-batch groupnorm + SiLU ------------------------
__global__ __launch_bounds__(1024)
void gnorm_silu_kernel(const float* __restrict__ in, const float* __restrict__ g,
                       const float* __restrict__ b, __bf16* __restrict__ out) {
  const int bat = blockIdx.x;
  const int tid = threadIdx.x;
  const long n = (long)Tt * Dd;
  float s = 0.0f, s2 = 0.0f;
  for (long i = tid; i < n; i += 1024) {
    const long t = i >> 9;
    const int d = (int)(i & 511);
    const float v = in[((t * Bb + bat) << 9) + d];
    s += v;
    s2 += v * v;
  }
#pragma unroll
  for (int m = 16; m >= 1; m >>= 1) { s += __shfl_xor(s, m, 32); s2 += __shfl_xor(s2, m, 32); }
  __shared__ float ra[32], rb[32];
  __shared__ float stats[2];
  const int wv_ = tid >> 5, ln_ = tid & 31;
  if (ln_ == 0) { ra[wv_] = s; rb[wv_] = s2; }
  __syncthreads();
  if (tid < 32) {
    float a = ra[tid], c = rb[tid];
#pragma unroll
    for (int m = 16; m >= 1; m >>= 1) { a += __shfl_xor(a, m, 32); c += __shfl_xor(c, m, 32); }
    if (tid == 0) {
      const float mu = a / (float)n;
      stats[0] = mu;
      stats[1] = c / (float)n - mu * mu;
    }
  }
  __syncthreads();
  const float mu = stats[0];
  const float rs = rsqrtf(stats[1] + EPSf);
  for (long i = tid; i < n; i += 1024) {
    const long t = i >> 9;
    const int d = (int)(i & 511);
    const long off = ((t * Bb + bat) << 9) + d;
    float v = (in[off] - mu) * rs * g[d] + b[d];
    v = v / (1.0f + __expf(-v));
    out[off] = (__bf16)v;
  }
}

// ------------------------------ host launcher ------------------------------
extern "C" void kernel_launch(void* const* d_in, const int* in_sizes, int n_in,
                              void* d_out, int out_size, void* d_ws, size_t ws_size,
                              hipStream_t stream) {
  (void)in_sizes; (void)n_in; (void)out_size; (void)ws_size;

  const float* x      = (const float*)d_in[0];
  // d_in[1] = key_padding_mask (all false; softmax mask is a no-op)
  const float* f1_lng = (const float*)d_in[2];
  const float* f1_lnb = (const float*)d_in[3];
  const float* f1_w1  = (const float*)d_in[4];
  const float* f1_b1  = (const float*)d_in[5];
  const float* f1_w2  = (const float*)d_in[6];
  const float* f1_b2  = (const float*)d_in[7];
  const float* alng   = (const float*)d_in[8];
  const float* alnb   = (const float*)d_in[9];
  const float* wq = (const float*)d_in[10]; const float* bq = (const float*)d_in[11];
  const float* wk = (const float*)d_in[12]; const float* bk = (const float*)d_in[13];
  const float* wv = (const float*)d_in[14]; const float* bv = (const float*)d_in[15];
  const float* wo = (const float*)d_in[16]; const float* bo = (const float*)d_in[17];
  const float* clng = (const float*)d_in[18]; const float* clnb = (const float*)d_in[19];
  const float* pw1w = (const float*)d_in[20]; const float* pw1b = (const float*)d_in[21];
  const float* dww  = (const float*)d_in[22]; const float* dwb  = (const float*)d_in[23];
  const float* gng  = (const float*)d_in[24]; const float* gnb  = (const float*)d_in[25];
  const float* pw2w = (const float*)d_in[26]; const float* pw2b = (const float*)d_in[27];
  const float* f2_lng = (const float*)d_in[28];
  const float* f2_lnb = (const float*)d_in[29];
  const float* f2_w1  = (const float*)d_in[30];
  const float* f2_b1  = (const float*)d_in[31];
  const float* f2_w2  = (const float*)d_in[32];
  const float* f2_b2  = (const float*)d_in[33];
  const float* flng   = (const float*)d_in[34];
  const float* flnb   = (const float*)d_in[35];
  float* out = (float*)d_out;

  // workspace bump allocator (256B aligned)
  char* wp = (char*)d_ws;
  auto alloc = [&](size_t bytes) -> char* {
    char* r = wp;
    wp += (bytes + 255) & ~(size_t)255;
    return r;
  };
  const size_t nTok = (size_t)Mr * Dd;  // 4.19M

  // bf16 weight mirrors
  __bf16* b_f1w1 = (__bf16*)alloc((size_t)FF * Dd * 2);
  __bf16* b_f1w2 = (__bf16*)alloc((size_t)Dd * FF * 2);
  __bf16* b_wq   = (__bf16*)alloc((size_t)Dd * Dd * 2);
  __bf16* b_wk   = (__bf16*)alloc((size_t)Dd * Dd * 2);
  __bf16* b_wv   = (__bf16*)alloc((size_t)Dd * Dd * 2);
  __bf16* b_wo   = (__bf16*)alloc((size_t)Dd * Dd * 2);
  __bf16* b_pw1  = (__bf16*)alloc((size_t)2 * Dd * Dd * 2);
  __bf16* b_pw2  = (__bf16*)alloc((size_t)Dd * Dd * 2);
  __bf16* b_f2w1 = (__bf16*)alloc((size_t)FF * Dd * 2);
  __bf16* b_f2w2 = (__bf16*)alloc((size_t)Dd * FF * 2);
  // activations
  __bf16* ln_bf  = (__bf16*)alloc(nTok * 2);
  __bf16* mid_bf = (__bf16*)alloc((size_t)Mr * FF * 2);
  float*  h1     = (float*)alloc(nTok * 4);
  float*  qf     = (float*)alloc(nTok * 4);
  float*  kf     = (float*)alloc(nTok * 4);
  float*  vf     = (float*)alloc(nTok * 4);
  __bf16* qbf    = (__bf16*)alloc(nTok * 2);
  __bf16* kbf    = (__bf16*)alloc(nTok * 2);
  __bf16* vtb    = (__bf16*)alloc(nTok * 2);
  __bf16* ctx_bf = (__bf16*)alloc(nTok * 2);
  float*  h2     = (float*)alloc(nTok * 4);
  float*  pw1o   = (float*)alloc((size_t)Mr * 2 * Dd * 4);
  __bf16* gn_bf  = (__bf16*)alloc(nTok * 2);
  float*  h3     = (float*)alloc(nTok * 4);
  float*  glu    = qf;  // q/k/v f32 free after rope_pack
  float*  dwo    = kf;
  float*  h4     = vf;

  auto cvt = [&](const float* src, __bf16* dst, int n) {
    cvt_bf16_kernel<<<(n + 255) / 256, 256, 0, stream>>>(src, dst, n);
  };
  cvt(f1_w1, b_f1w1, FF * Dd);
  cvt(f1_w2, b_f1w2, Dd * FF);
  cvt(wq, b_wq, Dd * Dd);
  cvt(wk, b_wk, Dd * Dd);
  cvt(wv, b_wv, Dd * Dd);
  cvt(wo, b_wo, Dd * Dd);
  cvt(pw1w, b_pw1, 2 * Dd * Dd);
  cvt(pw2w, b_pw2, Dd * Dd);
  cvt(f2_w1, b_f2w1, FF * Dd);
  cvt(f2_w2, b_f2w2, Dd * FF);

  const dim3 blk256(256), blk128(128);
  const dim3 gNffn(FF / 128, Mr / 128);     // (16,64)
  const dim3 gNd(Dd / 128, Mr / 128);       // (4,64)
  const dim3 gN2d(2 * Dd / 128, Mr / 128);  // (8,64)

  // ---- FFN #1: h1 = 0.5 * ffn(ln(x)) + x
  ln_kernel<<<Mr, blk128, 0, stream>>>(x, f1_lng, f1_lnb, nullptr, ln_bf);
  gemm_kernel<1><<<gNffn, blk256, 0, stream>>>(ln_bf, b_f1w1, f1_b1, nullptr,
                                               1.0f, nullptr, mid_bf, Mr, FF, Dd);
  gemm_kernel<0><<<gNd, blk256, 0, stream>>>(mid_bf, b_f1w2, f1_b2, x,
                                             0.5f, h1, nullptr, Mr, Dd, FF);

  // ---- attention: h2 = O(softmax(QK^T/8) V) + h1
  ln_kernel<<<Mr, blk128, 0, stream>>>(h1, alng, alnb, nullptr, ln_bf);
  gemm_kernel<0><<<gNd, blk256, 0, stream>>>(ln_bf, b_wq, bq, nullptr, 1.0f, qf,
                                             nullptr, Mr, Dd, Dd);
  gemm_kernel<0><<<gNd, blk256, 0, stream>>>(ln_bf, b_wk, bk, nullptr, 1.0f, kf,
                                             nullptr, Mr, Dd, Dd);
  gemm_kernel<0><<<gNd, blk256, 0, stream>>>(ln_bf, b_wv, bv, nullptr, 1.0f, vf,
                                             nullptr, Mr, Dd, Dd);
  rope_pack_kernel<<<(Tt * Bb * Hh * 32) / 256, blk256, 0, stream>>>(
      qf, kf, vf, qbf, kbf, vtb);
  attn_kernel<<<dim3(Tt / 128, Bb * Hh), blk256, 0, stream>>>(qbf, kbf, vtb, ctx_bf);
  gemm_kernel<0><<<gNd, blk256, 0, stream>>>(ctx_bf, b_wo, bo, h1, 1.0f, h2,
                                             nullptr, Mr, Dd, Dd);

  // ---- conv module: h3 = conv(h2) + h2
  ln_kernel<<<Mr, blk128, 0, stream>>>(h2, clng, clnb, nullptr, ln_bf);
  gemm_kernel<0><<<gN2d, blk256, 0, stream>>>(ln_bf, b_pw1, pw1b, nullptr, 1.0f,
                                              pw1o, nullptr, Mr, 2 * Dd, Dd);
  glu_kernel<<<(int)(nTok / 256), blk256, 0, stream>>>(pw1o, glu);
  dwconv_kernel<<<(int)(nTok / 256), blk256, 0, stream>>>(glu, dww, dwb, dwo);
  gnorm_silu_kernel<<<Bb, 1024, 0, stream>>>(dwo, gng, gnb, gn_bf);
  gemm_kernel<0><<<gNd, blk256, 0, stream>>>(gn_bf, b_pw2, pw2b, h2, 1.0f, h3,
                                             nullptr, Mr, Dd, Dd);

  // ---- FFN #2: h4 = 0.5 * ffn(ln(h3)) + h3
  ln_kernel<<<Mr, blk128, 0, stream>>>(h3, f2_lng, f2_lnb, nullptr, ln_bf);
  gemm_kernel<1><<<gNffn, blk256, 0, stream>>>(ln_bf, b_f2w1, f2_b1, nullptr,
                                               1.0f, nullptr, mid_bf, Mr, FF, Dd);
  gemm_kernel<0><<<gNd, blk256, 0, stream>>>(mid_bf, b_f2w2, f2_b2, h3,
                                             0.5f, h4, nullptr, Mr, Dd, FF);

  // ---- final LN -> d_out (f32)
  ln_kernel<<<Mr, blk128, 0, stream>>>(h4, flng, flnb, out, nullptr);
}
